// FusionLayer_25701084299870
// MI455X (gfx1250) — compile-verified, matrix operand order checked by reading, not run on hardware
//
#include <hip/hip_runtime.h>
#include <math.h>

// ---------------------------------------------------------------- types
typedef __attribute__((ext_vector_type(16))) __bf16       v16bf;
typedef __attribute__((ext_vector_type(8)))  float        v8f;
typedef __attribute__((ext_vector_type(4)))  unsigned int v4u;
typedef unsigned short u16;

union Frag {
    v16bf v;
    u16   s[16];
    v4u   q[2];
};

static __device__ inline u16 f2b(float f) {
    union { float f; unsigned u; } x; x.f = f;
    unsigned u = x.u;
    u += 0x7FFFu + ((u >> 16) & 1u);          // round to nearest even
    return (u16)(u >> 16);
}

// Load a 16x32 bf16 A/B fragment half for this lane.
// p must already point at  row_base + k0 + koff  (koff = 0 or 8 by lane>>4).
// Layout per ISA 7.12.2: lanes 0-15 hold K {0..7,16..23}, lanes 16-31 {8..15,24..31}.
static __device__ inline v16bf ldfrag16(const u16* p) {
    Frag f;
    f.q[0] = *(const v4u*)(p);
    f.q[1] = *(const v4u*)(p + 16);
    return f.v;
}

static __device__ inline v8f wmma_bf16(v16bf a, v16bf b, v8f c) {
    return __builtin_amdgcn_wmma_f32_16x16x32_bf16(
        /*neg_a=*/false, a, /*neg_b=*/false, b,
        /*c_mod=*/(short)0, c, /*reuse_a=*/false, /*reuse_b=*/false);
}

#define NTOT   12160
#define EMB    512
#define ESM    1280
#define NHEAD  8
#define HD     64
#define NGRAPH 32

// ---------------------------------------------------------------- fp32 -> bf16
__global__ void cvt_bf16_kernel(const float* __restrict__ src,
                                u16* __restrict__ dst, int n) {
    int i = blockIdx.x * blockDim.x + threadIdx.x;
    if (i < n) dst[i] = f2b(src[i]);
}

// ---------------------------------------------------------------- GEMM  C = A * W^T + bias
// A:[M,K] bf16 row-major, W:[Nc,K] bf16 row-major (weight), bias fp32 [Nc].
// One wave computes a 64x64 tile of C with 4x4 = 16 WMMA accumulators.
// launch_bounds(128, 1): ~220 live VGPRs must NOT spill -> request full register file.
__global__ void __launch_bounds__(128, 1)
gemm_bf16_kernel(const u16* __restrict__ A,
                 const u16* __restrict__ W,
                 const float* __restrict__ bias,
                 int M, int Nc, int Kc,
                 u16* __restrict__ Obf, float* __restrict__ Of) {
    const int wave   = blockIdx.x * (blockDim.x >> 5) + (threadIdx.x >> 5);
    const int tilesN = Nc >> 6;
    const int tilesM = M >> 6;
    if (wave >= tilesM * tilesN) return;               // wave-uniform
    const int tm   = (wave / tilesN) << 6;
    const int tn   = (wave % tilesN) << 6;
    const int lane = threadIdx.x & 31;
    const int nl   = lane & 15;
    const int koff = (lane & 16) ? 8 : 0;

    const v8f zero = {0.f,0.f,0.f,0.f,0.f,0.f,0.f,0.f};
    v8f acc[4][4];
#pragma unroll
    for (int i = 0; i < 4; ++i)
#pragma unroll
        for (int j = 0; j < 4; ++j) acc[i][j] = zero;

    const u16* arow[4];
    const u16* wrow[4];
#pragma unroll
    for (int i = 0; i < 4; ++i) arow[i] = A + (size_t)(tm + 16 * i + nl) * Kc;
#pragma unroll
    for (int j = 0; j < 4; ++j) wrow[j] = W + (size_t)(tn + 16 * j + nl) * Kc;

    for (int k0 = 0; k0 < Kc; k0 += 32) {
        v16bf a[4], b[4];
#pragma unroll
        for (int i = 0; i < 4; ++i) {
            a[i] = ldfrag16(arow[i] + k0 + koff);
            __builtin_prefetch(arow[i] + k0 + 256, 0, 3);   // near-scope prefetch
        }
#pragma unroll
        for (int j = 0; j < 4; ++j) b[j] = ldfrag16(wrow[j] + k0 + koff);
#pragma unroll
        for (int i = 0; i < 4; ++i)
#pragma unroll
            for (int j = 0; j < 4; ++j)
                acc[i][j] = wmma_bf16(a[i], b[j], acc[i][j]);
    }

    // C layout: VGPR r -> M = r + 8*(lane>=16), N = lane&15 (per 16x16 tile)
    const int mhi = (lane >> 4) << 3;
#pragma unroll
    for (int i = 0; i < 4; ++i)
#pragma unroll
        for (int j = 0; j < 4; ++j)
#pragma unroll
            for (int r = 0; r < 8; ++r) {
                const int m = tm + 16 * i + r + mhi;
                const int n = tn + 16 * j + nl;
                const float val = acc[i][j][r] + bias[n];
                if (Of) Of[(size_t)m * Nc + n] = val;
                else    Obf[(size_t)m * Nc + n] = f2b(val);
            }
}

// ---------------------------------------------------------------- segment attention
// One wave per (graph, head, 16-row q tile). Q,K,V: [NTOT, 512] bf16.
__global__ void __launch_bounds__(32, 1)
attn_kernel(const u16* __restrict__ Q,
            const u16* __restrict__ K,
            const u16* __restrict__ V,
            u16* __restrict__ O) {
    const int qt   = blockIdx.x;
    const int head = blockIdx.y;
    const int b    = blockIdx.z;
    const int L    = 256 + 8 * b;
    if (qt * 16 >= L) return;                          // wave-uniform
    const int start = 256 * b + 4 * b * (b - 1);
    const int Lp    = (L + 31) & ~31;                  // pad to WMMA K=32

    __shared__ float sc[16][512];
    __shared__ u16   pb[16][512];
    __shared__ float mrow[16], srow[16];

    const int lane = threadIdx.x & 31;
    const int nl   = lane & 15;
    const int hi   = lane >> 4;
    const int koff = hi ? 8 : 0;
    const int mhi  = hi << 3;

    // ---- Q fragments (16 rows x 64 head dims -> two K=32 fragments)
    int qrow = start + qt * 16 + nl;
    if (qrow > NTOT - 1) qrow = NTOT - 1;
    const u16* qp = Q + (size_t)qrow * EMB + head * HD;
    const v16bf qa0 = ldfrag16(qp + koff);
    const v16bf qa1 = ldfrag16(qp + 32 + koff);

    // ---- scores = Q K^T / 8 into LDS
    const int nkt16 = Lp >> 4;
    for (int kt = 0; kt < nkt16; ++kt) {
        int krow = start + kt * 16 + nl;
        if (krow > NTOT - 1) krow = NTOT - 1;
        const u16* kp = K + (size_t)krow * EMB + head * HD;
        const v16bf kb0 = ldfrag16(kp + koff);
        const v16bf kb1 = ldfrag16(kp + 32 + koff);
        v8f acc = {0.f,0.f,0.f,0.f,0.f,0.f,0.f,0.f};
        acc = wmma_bf16(qa0, kb0, acc);
        acc = wmma_bf16(qa1, kb1, acc);
#pragma unroll
        for (int r = 0; r < 8; ++r)
            sc[r + mhi][kt * 16 + nl] = acc[r] * 0.125f;   // 1/sqrt(64)
    }
    // mask pad keys
    for (int c = L + lane; c < Lp; c += 32)
        for (int r = 0; r < 16; ++r) sc[r][c] = -INFINITY;
    __syncthreads();

    // ---- fp32 softmax, row per lane (lanes 0-15)
    if (lane < 16) {
        float m = -INFINITY;
        for (int l = 0; l < Lp; ++l) m = fmaxf(m, sc[lane][l]);
        float s = 0.f;
        for (int l = 0; l < Lp; ++l) s += __expf(sc[lane][l] - m);
        mrow[lane] = m;
        srow[lane] = 1.f / s;
    }
    __syncthreads();
    for (int idx = lane; idx < 16 * Lp; idx += 32) {
        const int r = idx / Lp;
        const int l = idx - r * Lp;
        pb[r][l] = f2b(__expf(sc[r][l] - mrow[r]) * srow[r]);  // pads -> 0
    }
    __syncthreads();

    // ---- out = P V   (A fragment from LDS, B gathered column-major from V)
    v8f oacc[4];
#pragma unroll
    for (int j = 0; j < 4; ++j) oacc[j] = (v8f){0.f,0.f,0.f,0.f,0.f,0.f,0.f,0.f};

    const int nkt32 = Lp >> 5;
    for (int kt = 0; kt < nkt32; ++kt) {
        const v16bf pa = ldfrag16(&pb[nl][kt * 32 + koff]);
        const int rowbase = start + kt * 32 + koff;
#pragma unroll
        for (int j = 0; j < 4; ++j) {
            const int col = head * HD + j * 16 + nl;
            Frag fb;
#pragma unroll
            for (int i = 0; i < 8; ++i) {
                int r0 = rowbase + i;      if (r0 > NTOT - 1) r0 = NTOT - 1;
                int r1 = rowbase + 16 + i; if (r1 > NTOT - 1) r1 = NTOT - 1;
                fb.s[i]     = V[(size_t)r0 * EMB + col];
                fb.s[8 + i] = V[(size_t)r1 * EMB + col];
            }
            oacc[j] = wmma_bf16(pa, fb.v, oacc[j]);
        }
    }
#pragma unroll
    for (int j = 0; j < 4; ++j)
#pragma unroll
        for (int r = 0; r < 8; ++r) {
            const int ml = qt * 16 + r + mhi;
            if (ml < L)
                O[(size_t)(start + ml) * EMB + head * HD + j * 16 + nl] =
                    f2b(oacc[j][r]);
        }
}

// ---------------------------------------------------------------- residual + LayerNorm
__global__ void ln_kernel(const float* __restrict__ o,
                          const float* __restrict__ resid,
                          const float* __restrict__ gamma,
                          const float* __restrict__ beta,
                          float* __restrict__ out) {
    const int row = blockIdx.x;
    const int t   = threadIdx.x;
    const size_t base = (size_t)row * EMB;
    const float x0 = o[base + t]       + resid[base + t];
    const float x1 = o[base + 256 + t] + resid[base + 256 + t];
    __shared__ float s1[256], s2[256];
    s1[t] = x0 + x1;
    s2[t] = x0 * x0 + x1 * x1;
    __syncthreads();
    for (int s = 128; s > 0; s >>= 1) {
        if (t < s) { s1[t] += s1[t + s]; s2[t] += s2[t + s]; }
        __syncthreads();
    }
    const float mean = s1[0] * (1.f / EMB);
    const float var  = s2[0] * (1.f / EMB) - mean * mean;
    const float inv  = rsqrtf(var + 1e-5f);
    out[base + t]       = (x0 - mean) * inv * gamma[t]       + beta[t];
    out[base + 256 + t] = (x1 - mean) * inv * gamma[256 + t] + beta[256 + t];
}

// ---------------------------------------------------------------- launch
extern "C" void kernel_launch(void* const* d_in, const int* in_sizes, int n_in,
                              void* d_out, int out_size, void* d_ws, size_t ws_size,
                              hipStream_t stream) {
    const float* esm   = (const float*)d_in[0];
    const float* h     = (const float*)d_in[1];
    const float* Wesm  = (const float*)d_in[2];
    const float* besm  = (const float*)d_in[3];
    const float* Wq    = (const float*)d_in[4];
    const float* bq    = (const float*)d_in[5];
    const float* Wk    = (const float*)d_in[6];
    const float* bk    = (const float*)d_in[7];
    const float* Wv    = (const float*)d_in[8];
    const float* bv    = (const float*)d_in[9];
    const float* Wo    = (const float*)d_in[10];
    const float* bo    = (const float*)d_in[11];
    const float* gamma = (const float*)d_in[12];
    const float* beta  = (const float*)d_in[13];
    float* out = (float*)d_out;

    // workspace layout
    char* w = (char*)d_ws;
    u16* esm_bf  = (u16*)w;                       w += (size_t)NTOT * ESM * 2;
    u16* h_bf    = (u16*)w;                       w += (size_t)NTOT * EMB * 2;
    u16* hesm_bf = (u16*)w;                       w += (size_t)NTOT * EMB * 2;
    u16* q_bf    = (u16*)w;                       w += (size_t)NTOT * EMB * 2;
    u16* k_bf    = (u16*)w;                       w += (size_t)NTOT * EMB * 2;
    u16* v_bf    = (u16*)w;                       w += (size_t)NTOT * EMB * 2;
    u16* attn_bf = (u16*)w;                       w += (size_t)NTOT * EMB * 2;
    float* o_f32 = (float*)w;                     w += (size_t)NTOT * EMB * 4;
    u16* Wesm_bf = (u16*)w;                       w += (size_t)EMB * ESM * 2;
    u16* Wq_bf   = (u16*)w;                       w += (size_t)EMB * EMB * 2;
    u16* Wk_bf   = (u16*)w;                       w += (size_t)EMB * EMB * 2;
    u16* Wv_bf   = (u16*)w;                       w += (size_t)EMB * EMB * 2;
    u16* Wo_bf   = (u16*)w;                       w += (size_t)EMB * EMB * 2;

    auto cvt = [&](const float* s, u16* d, int n) {
        cvt_bf16_kernel<<<(n + 255) / 256, 256, 0, stream>>>(s, d, n);
    };
    cvt(esm,  esm_bf,  NTOT * ESM);
    cvt(h,    h_bf,    NTOT * EMB);
    cvt(Wesm, Wesm_bf, EMB * ESM);
    cvt(Wq,   Wq_bf,   EMB * EMB);
    cvt(Wk,   Wk_bf,   EMB * EMB);
    cvt(Wv,   Wv_bf,   EMB * EMB);
    cvt(Wo,   Wo_bf,   EMB * EMB);

    const int tiles = (NTOT / 64) * (EMB / 64);   // 1520 wave tiles
    const int gblk  = (tiles + 3) / 4;            // 4 waves / block

    // h_esm = esm @ Wesm^T + besm   (bf16 out)
    gemm_bf16_kernel<<<gblk, 128, 0, stream>>>(esm_bf, Wesm_bf, besm,
                                               NTOT, EMB, ESM, hesm_bf, nullptr);
    // q = h @ Wq^T + bq ; k/v from h_esm
    gemm_bf16_kernel<<<gblk, 128, 0, stream>>>(h_bf,    Wq_bf, bq,
                                               NTOT, EMB, EMB, q_bf, nullptr);
    gemm_bf16_kernel<<<gblk, 128, 0, stream>>>(hesm_bf, Wk_bf, bk,
                                               NTOT, EMB, EMB, k_bf, nullptr);
    gemm_bf16_kernel<<<gblk, 128, 0, stream>>>(hesm_bf, Wv_bf, bv,
                                               NTOT, EMB, EMB, v_bf, nullptr);

    // segment attention
    attn_kernel<<<dim3(32, NHEAD, NGRAPH), 32, 0, stream>>>(q_bf, k_bf, v_bf, attn_bf);

    // out = attn @ Wo^T + bo  (fp32 out)
    gemm_bf16_kernel<<<gblk, 128, 0, stream>>>(attn_bf, Wo_bf, bo,
                                               NTOT, EMB, EMB, nullptr, o_f32);

    // residual + layernorm
    ln_kernel<<<NTOT, 256, 0, stream>>>(o_f32, h, gamma, beta, out);
}
